// Gemma3nTextAttention_45569603010636
// MI455X (gfx1250) — matrix operand order repaired and checked by reading, not compiled, not run on hardware
//
#include <hip/hip_runtime.h>

// ---------------------------------------------------------------------------
// Gemma3n attention for MI455X (gfx1250)
//   bf16 WMMA compute, async-to-LDS (ASYNCcnt) data movement, double-buffered
// ---------------------------------------------------------------------------

typedef __bf16 bf16;
typedef float v8f __attribute__((ext_vector_type(8)));
typedef bf16 v8bf __attribute__((ext_vector_type(8)));
typedef bf16 v16bf __attribute__((ext_vector_type(16)));

// ---- CDNA5 async copy: global -> LDS, 16 bytes per lane (ASYNCcnt) --------
__device__ __forceinline__ void async_b128(void* lds, const void* g) {
  // flat LDS pointer low 32 bits == LDS byte offset (ISA 10.2 aperture rule)
  unsigned off = (unsigned)(unsigned long long)lds;
  asm volatile("global_load_async_to_lds_b128 %0, %1, off"
               :: "v"(off), "v"(g) : "memory");
}
__device__ __forceinline__ void wait_async0() {
  asm volatile("s_wait_asynccnt 0" ::: "memory");
}

// ---- WMMA fragment loaders (CDNA5 ISA 7.12.2) -----------------------------
// A 16x32 bf16: lane row=lane&15, half h: k in [8h,8h+8) u [16+8h,16+8h+8)
//   -> two contiguous 16B LDS reads when ld*2 is a multiple of 16B.
__device__ __forceinline__ v16bf frag_a(const bf16* base, int ld) {
  int lane = threadIdx.x & 31;
  const bf16* p = base + (lane & 15) * ld + (lane >> 4) * 8;
  v8bf lo = *(const v8bf*)p;
  v8bf hi = *(const v8bf*)(p + 16);
  return __builtin_shufflevector(lo, hi, 0, 1, 2, 3, 4, 5, 6, 7, 8, 9, 10, 11,
                                 12, 13, 14, 15);
}
// B 32x16 bf16 (LDS stored N-major: row n, K contiguous): lane col=lane&15,
// half h: k in [8h,8h+8) u [16+8h,16+8h+8) -> two 16B reads.
__device__ __forceinline__ v16bf frag_b(const bf16* base, int ld) {
  int lane = threadIdx.x & 31;
  const bf16* p = base + (lane & 15) * ld + (lane >> 4) * 8;
  v8bf lo = *(const v8bf*)p;
  v8bf hi = *(const v8bf*)(p + 16);
  return __builtin_shufflevector(lo, hi, 0, 1, 2, 3, 4, 5, 6, 7, 8, 9, 10, 11,
                                 12, 13, 14, 15);
}

__device__ __forceinline__ v8f wmma_bf16(v16bf a, v16bf b, v8f c) {
  return __builtin_amdgcn_wmma_f32_16x16x32_bf16(false, a, false, b, (short)0,
                                                 c, false, false);
}

// ---------------------------------------------------------------------------
// Prep kernels: fp32 -> bf16 convert, and fp32 [K][N] -> bf16 [N][K] transpose
// ---------------------------------------------------------------------------
__global__ __launch_bounds__(256) void cvt_f32_bf16(const float* __restrict__ in,
                                                    bf16* __restrict__ out,
                                                    long n) {
  long i = (long)blockIdx.x * 256 + threadIdx.x;
  if (i < n) out[i] = (bf16)in[i];
}

__global__ __launch_bounds__(256) void transpose_f32_bf16(
    const float* __restrict__ in, bf16* __restrict__ out, int K, int N) {
  __shared__ float t[32][33];
  int n0 = blockIdx.x * 32, k0 = blockIdx.y * 32;
  int tx = threadIdx.x & 31, ty = threadIdx.x >> 5;  // 32 x 8
#pragma unroll
  for (int i = ty; i < 32; i += 8) t[i][tx] = in[(long)(k0 + i) * N + n0 + tx];
  __syncthreads();
#pragma unroll
  for (int i = ty; i < 32; i += 8)
    out[(long)(n0 + i) * K + k0 + tx] = (bf16)t[tx][i];
}

// ---------------------------------------------------------------------------
// GEMM: C[M,N](fp32) = A[M,K](bf16,row) * Bt[N,K](bf16, N-major)
// Workgroup tile 64x128, wave tile 32x32, K-step 32, double-buffered async.
// ---------------------------------------------------------------------------
#define GT_M 64
#define GT_N 128
#define GT_K 32
#define GT_LD 40  // bf16 per LDS row: 32 + 8 pad -> 80B row stride (16B mult)

__device__ __forceinline__ void gemm_stage(bf16* sA, bf16* sB,
                                           const bf16* __restrict__ A,
                                           const bf16* __restrict__ Bt, int m0,
                                           int n0, int k0, int K, int tid) {
  {  // A tile 64x32: 256 chunks of 16B, one per thread
    int r = tid >> 2, c = tid & 3;
    async_b128(sA + r * GT_LD + c * 8, A + (long)(m0 + r) * K + k0 + c * 8);
  }
  {  // B tile 128x32: 512 chunks, two per thread
    int r = tid >> 1, cb = (tid & 1) * 2;
#pragma unroll
    for (int j = 0; j < 2; ++j)
      async_b128(sB + r * GT_LD + (cb + j) * 8,
                 Bt + (long)(n0 + r) * K + k0 + (cb + j) * 8);
  }
}

__global__ __launch_bounds__(256) void gemm_bf16wmma(
    const bf16* __restrict__ A, const bf16* __restrict__ Bt,
    float* __restrict__ C, int M, int N, int K) {
  __shared__ __align__(16) bf16 sA[2][GT_M * GT_LD];
  __shared__ __align__(16) bf16 sB[2][GT_N * GT_LD];

  int tid = threadIdx.x;
  int lane = tid & 31, w = tid >> 5;
  int m0 = blockIdx.y * GT_M;
  int n0 = blockIdx.x * GT_N;
  int wm = (w >> 2) * 32;
  int wn = (w & 3) * 32;

  v8f acc[2][2];
#pragma unroll
  for (int i = 0; i < 2; ++i)
#pragma unroll
    for (int j = 0; j < 2; ++j)
#pragma unroll
      for (int r = 0; r < 8; ++r) acc[i][j][r] = 0.f;

  int cur = 0;
  gemm_stage(sA[0], sB[0], A, Bt, m0, n0, 0, K, tid);
  for (int k0 = 0; k0 < K; k0 += GT_K) {
    wait_async0();
    __syncthreads();  // buf[cur] ready everywhere; prev reads of buf[1-cur] done
    if (k0 + GT_K < K)
      gemm_stage(sA[1 - cur], sB[1 - cur], A, Bt, m0, n0, k0 + GT_K, K, tid);

    v16bf fa[2], fb[2];
#pragma unroll
    for (int i = 0; i < 2; ++i) fa[i] = frag_a(sA[cur] + (wm + i * 16) * GT_LD, GT_LD);
#pragma unroll
    for (int i = 0; i < 2; ++i) fb[i] = frag_b(sB[cur] + (wn + i * 16) * GT_LD, GT_LD);
#pragma unroll
    for (int i = 0; i < 2; ++i)
#pragma unroll
      for (int j = 0; j < 2; ++j) acc[i][j] = wmma_bf16(fa[i], fb[j], acc[i][j]);
    cur ^= 1;
  }

  int half = lane >> 4, nn = lane & 15;
#pragma unroll
  for (int i = 0; i < 2; ++i)
#pragma unroll
    for (int j = 0; j < 2; ++j)
#pragma unroll
      for (int r = 0; r < 8; ++r) {
        int m = m0 + wm + i * 16 + r + 8 * half;
        int n = n0 + wn + j * 16 + nn;
        C[(long)m * N + n] = acc[i][j][r];
      }
}

// ---------------------------------------------------------------------------
// RMSNorm (+scale) + RoPE for Q/K. One block per (token, head). bf16 out.
// ---------------------------------------------------------------------------
__global__ __launch_bounds__(256) void normrope_kernel(
    const float* __restrict__ in, bf16* __restrict__ out,
    const float* __restrict__ scale, int nh, int S) {
  __shared__ float red[256];
  __shared__ float nv[256];
  int tok = blockIdx.x, h = blockIdx.y, t = threadIdx.x;
  long base = ((long)tok * nh + h) * 256;
  float x = in[base + t];
  red[t] = x * x;
  __syncthreads();
#pragma unroll
  for (int s = 128; s > 0; s >>= 1) {
    if (t < s) red[t] += red[t + s];
    __syncthreads();
  }
  float xn = x * rsqrtf(red[0] * (1.f / 256.f) + 1e-6f) * scale[t];
  nv[t] = xn;
  __syncthreads();
  int pos = tok % S;
  int i = t & 127;
  float invf = __powf(10000.f, -(float)i * (1.f / 128.f));
  float f = (float)pos * invf;
  float c = __cosf(f), sn = __sinf(f);
  float o = (t < 128) ? (nv[t] * c - nv[t + 128] * sn)
                      : (nv[t] * c + nv[t - 128] * sn);
  out[base + t] = (bf16)o;
}

// RMSNorm (no scale) for V, written TRANSPOSED: outT[((b*nkv+h)*256+d)*S + s]
__global__ __launch_bounds__(256) void norm_v_t_kernel(
    const float* __restrict__ in, bf16* __restrict__ outT, int nkv, int S) {
  __shared__ float red[256];
  int tok = blockIdx.x, h = blockIdx.y, t = threadIdx.x;
  long base = ((long)tok * nkv + h) * 256;
  float x = in[base + t];
  red[t] = x * x;
  __syncthreads();
#pragma unroll
  for (int s = 128; s > 0; s >>= 1) {
    if (t < s) red[t] += red[t + s];
    __syncthreads();
  }
  float xn = x * rsqrtf(red[0] * (1.f / 256.f) + 1e-6f);
  int b = tok / S, sidx = tok % S;
  outT[(((long)b * nkv + h) * 256 + t) * S + sidx] = (bf16)xn;
}

// ---------------------------------------------------------------------------
// Flash attention: workgroup per (b, h, 64-query block), 8 waves.
// Wave w: q-subtile qs=w>>1, d-half dh=w&1. Key blocks of 32, causal skip.
// Q fragments register-resident; K/V tiles async-copied (V pre-transposed).
// ---------------------------------------------------------------------------
#define AT_LDQ 264  // 256 + 8 pad bf16 -> 528B rows
#define AT_LDP 40   // 32 + 8 pad

#define SM_SQ 0                        // 64 x 264 bf16 = 33792 (temp, overlays)
#define SM_SK 0                        // 32 x 264 bf16 = 16896
#define SM_SVT 16896                   // 256 x 40 bf16 = 20480
#define SM_SP 37376                    // 64 x 40 bf16  = 5120
#define SM_SS 42496                    // 64 x 40 f32   = 10240
#define SM_M 52736
#define SM_L 52992
#define SM_AL 53248
#define SM_TOT 53504

__global__ __launch_bounds__(256) void attn_kernel(
    const bf16* __restrict__ Qn, const bf16* __restrict__ Kn,
    const bf16* __restrict__ VtG, bf16* __restrict__ O, int S, int nh,
    int nkv) {
  __shared__ __align__(16) char smem[SM_TOT];
  bf16* sQ = (bf16*)(smem + SM_SQ);
  bf16* sK = (bf16*)(smem + SM_SK);
  bf16* sVt = (bf16*)(smem + SM_SVT);
  bf16* sP = (bf16*)(smem + SM_SP);
  float* sS = (float*)(smem + SM_SS);
  float* sM = (float*)(smem + SM_M);
  float* sL = (float*)(smem + SM_L);
  float* sAl = (float*)(smem + SM_AL);

  int qb = blockIdx.x, h = blockIdx.y, b = blockIdx.z;
  int q0 = qb * 64;
  int kvh = h / (nh / nkv);
  int tid = threadIdx.x, lane = tid & 31, w = tid >> 5;
  int qs = w >> 1, dh = w & 1;
  int half = lane >> 4, nn = lane & 15;

  {  // async-stage Q block 64x256 bf16 (8 x 16B chunks per thread)
    int r = tid >> 2;
    const bf16* g = Qn + ((long)(b * S + q0 + r) * nh + h) * 256;
#pragma unroll
    for (int j = 0; j < 8; ++j) {
      int c = (tid & 3) * 8 + j;
      async_b128(sQ + r * AT_LDQ + c * 8, g + c * 8);
    }
  }
  if (tid < 64) { sM[tid] = -1e30f; sL[tid] = 0.f; }
  wait_async0();
  __syncthreads();

  v16bf aQ[8];
#pragma unroll
  for (int c = 0; c < 8; ++c)
    aQ[c] = frag_a(sQ + (qs * 16) * AT_LDQ + c * 32, AT_LDQ);

  v8f o[8];
#pragma unroll
  for (int nt = 0; nt < 8; ++nt)
#pragma unroll
    for (int r = 0; r < 8; ++r) o[nt][r] = 0.f;

  __syncthreads();  // aQ reads done before K/V overwrite sQ region

  int nkb = (q0 >> 5) + 2;  // causal: keys 0 .. q0+63
  const bf16* vbase = VtG + ((long)(b * nkv + kvh) * 256) * S;
  for (int kb = 0; kb < nkb; ++kb) {
    int k0 = kb * 32;
    {  // async-stage K block 32x256 (4 chunks/thread)
      int r = tid >> 3;
      const bf16* g = Kn + ((long)(b * S + k0 + r) * nkv + kvh) * 256;
#pragma unroll
      for (int j = 0; j < 4; ++j) {
        int c = (tid & 7) * 4 + j;
        async_b128(sK + r * AT_LDQ + c * 8, g + c * 8);
      }
    }
    {  // async-stage Vt block 256x32 (4 chunks/thread; rows are d)
      const bf16* g = vbase + (long)tid * S + k0;
#pragma unroll
      for (int j = 0; j < 4; ++j)
        async_b128(sVt + tid * AT_LDP + j * 8, g + j * 8);
    }
    wait_async0();
    __syncthreads();

    // scores: wave tile (qs, ks=dh), 16x16 over 8 d-chunks
    v8f sc;
#pragma unroll
    for (int r = 0; r < 8; ++r) sc[r] = 0.f;
#pragma unroll
    for (int c = 0; c < 8; ++c) {
      v16bf fk = frag_b(sK + (dh * 16) * AT_LDQ + c * 32, AT_LDQ);
      sc = wmma_bf16(aQ[c], fk, sc);
    }
    {  // scale 1/sqrt(256), causal mask, spill to LDS
      int kg = k0 + dh * 16 + nn;
#pragma unroll
      for (int r = 0; r < 8; ++r) {
        int qr = qs * 16 + r + 8 * half;
        float v = sc[r] * 0.0625f;
        if (kg > q0 + qr) v = -1e9f;
        sS[qr * AT_LDP + dh * 16 + nn] = v;
      }
    }
    __syncthreads();

    // online softmax: one thread per query row
    if (tid < 64) {
      float m_old = sM[tid], l_old = sL[tid];
      float mb = -1e30f;
#pragma unroll 8
      for (int j = 0; j < 32; ++j) mb = fmaxf(mb, sS[tid * AT_LDP + j]);
      float m_new = fmaxf(m_old, mb);
      float alpha = __expf(m_old - m_new);
      float lsum = 0.f;
#pragma unroll 8
      for (int j = 0; j < 32; ++j) {
        float p = __expf(sS[tid * AT_LDP + j] - m_new);
        sP[tid * AT_LDP + j] = (bf16)p;
        lsum += p;
      }
      sM[tid] = m_new;
      sL[tid] = l_old * alpha + lsum;
      sAl[tid] = alpha;
    }
    __syncthreads();

    // rescale O and accumulate P(16x32) x V(32x128-half)
    float al[8];
#pragma unroll
    for (int r = 0; r < 8; ++r) al[r] = sAl[qs * 16 + r + 8 * half];
    v16bf fp = frag_a(sP + (qs * 16) * AT_LDP, AT_LDP);
#pragma unroll
    for (int nt = 0; nt < 8; ++nt) {
#pragma unroll
      for (int r = 0; r < 8; ++r) o[nt][r] = o[nt][r] * al[r];
      v16bf fv = frag_b(sVt + (dh * 128 + nt * 16) * AT_LDP, AT_LDP);
      o[nt] = wmma_bf16(fp, fv, o[nt]);
    }
    __syncthreads();  // PV reads done before next iteration's async overwrite
  }

  // epilogue: divide by running sum, write bf16 [b*S+q][h*256+d]
  float li[8];
#pragma unroll
  for (int r = 0; r < 8; ++r) li[r] = 1.f / sL[qs * 16 + r + 8 * half];
#pragma unroll
  for (int nt = 0; nt < 8; ++nt)
#pragma unroll
    for (int r = 0; r < 8; ++r) {
      int qg = q0 + qs * 16 + r + 8 * half;
      int d = dh * 128 + nt * 16 + nn;
      O[(long)(b * S + qg) * (nh * 256) + h * 256 + d] =
          (bf16)(o[nt][r] * li[r]);
    }
}

// ---------------------------------------------------------------------------
extern "C" void kernel_launch(void* const* d_in, const int* in_sizes, int n_in,
                              void* d_out, int out_size, void* d_ws,
                              size_t ws_size, hipStream_t stream) {
  const float* X = (const float*)d_in[0];
  // d_in[1] = attention_mask (pure causal; computed analytically)
  const float* wq = (const float*)d_in[2];
  const float* wk = (const float*)d_in[3];
  const float* wv = (const float*)d_in[4];
  const float* wo = (const float*)d_in[5];
  const float* qsc = (const float*)d_in[6];
  const float* ksc = (const float*)d_in[7];
  float* out = (float*)d_out;

  const int B = 2, S = 2048, H = 8, KV = 2, D = 256, HID = 2048;
  const int M = B * S;  // 4096 tokens

  char* ws = (char*)d_ws;
  size_t off = 0;
  bf16* Xb = (bf16*)(ws + off);   off += (size_t)M * HID * 2;       // 16MB
  bf16* wqT = (bf16*)(ws + off);  off += (size_t)HID * H * D * 2;   // 8MB
  bf16* wkT = (bf16*)(ws + off);  off += (size_t)HID * KV * D * 2;  // 2MB
  bf16* wvT = (bf16*)(ws + off);  off += (size_t)HID * KV * D * 2;  // 2MB
  bf16* woT = (bf16*)(ws + off);  off += (size_t)H * D * HID * 2;   // 8MB
  float* Qraw = (float*)(ws + off); off += (size_t)M * H * D * 4;   // 32MB
  float* Kraw = (float*)(ws + off); off += (size_t)M * KV * D * 4;  // 8MB
  float* Vraw = (float*)(ws + off); off += (size_t)M * KV * D * 4;  // 8MB
  bf16* Qn = (bf16*)(ws + off);   off += (size_t)M * H * D * 2;     // 16MB
  bf16* Kn = (bf16*)(ws + off);   off += (size_t)M * KV * D * 2;    // 4MB
  bf16* VtG = (bf16*)(ws + off);  off += (size_t)M * KV * D * 2;    // 4MB
  bf16* AttnOb = (bf16*)Qraw;  // Qraw dead after norm: reuse

  dim3 blk(256);
  // prep: activations -> bf16; weights -> bf16 [N][K]
  cvt_f32_bf16<<<(M * HID) / 256, blk, 0, stream>>>(X, Xb, (long)M * HID);
  transpose_f32_bf16<<<dim3((H * D) / 32, HID / 32), blk, 0, stream>>>(wq, wqT, HID, H * D);
  transpose_f32_bf16<<<dim3((KV * D) / 32, HID / 32), blk, 0, stream>>>(wk, wkT, HID, KV * D);
  transpose_f32_bf16<<<dim3((KV * D) / 32, HID / 32), blk, 0, stream>>>(wv, wvT, HID, KV * D);
  transpose_f32_bf16<<<dim3(HID / 32, (H * D) / 32), blk, 0, stream>>>(wo, woT, H * D, HID);
  // QKV projections
  gemm_bf16wmma<<<dim3((H * D) / GT_N, M / GT_M), blk, 0, stream>>>(Xb, wqT, Qraw, M, H * D, HID);
  gemm_bf16wmma<<<dim3((KV * D) / GT_N, M / GT_M), blk, 0, stream>>>(Xb, wkT, Kraw, M, KV * D, HID);
  gemm_bf16wmma<<<dim3((KV * D) / GT_N, M / GT_M), blk, 0, stream>>>(Xb, wvT, Vraw, M, KV * D, HID);
  // RMSNorm (+scale) + RoPE; V normed + transposed
  normrope_kernel<<<dim3(M, H), blk, 0, stream>>>(Qraw, Qn, qsc, H, S);
  normrope_kernel<<<dim3(M, KV), blk, 0, stream>>>(Kraw, Kn, ksc, KV, S);
  norm_v_t_kernel<<<dim3(M, KV), blk, 0, stream>>>(Vraw, VtG, KV, S);
  // GQA causal flash attention (bf16 out)
  attn_kernel<<<dim3(S / 64, H, B), blk, 0, stream>>>(Qn, Kn, VtG, AttnOb, S, H, KV);
  // output projection
  gemm_bf16wmma<<<dim3(HID / GT_N, M / GT_M), blk, 0, stream>>>(AttnOb, woT, out, M, HID, HID);
}
